// Upsamplerx16_10565619548893
// MI455X (gfx1250) — matrix operand order repaired and checked by reading, not compiled, not run on hardware
//
#include <hip/hip_runtime.h>
#include <hip/hip_bf16.h>

typedef _Float16 v16h __attribute__((ext_vector_type(16)));
typedef _Float16 v8h  __attribute__((ext_vector_type(8)));
typedef float    v8f  __attribute__((ext_vector_type(8)));

#define N_IMG    4
#define C_IN     128
#define HW       4096            // 64*64
#define MPIX     16384           // 4*64*64 flat pixels
#define C_M      256             // UP*UP
#define C_MASK   2304            // UP*UP*9
#define K_I2C    1152            // C_IN*9

static __device__ __forceinline__ v16h pack16(v8h lo, v8h hi){
  v16h r;
#pragma unroll
  for (int i = 0; i < 8; ++i){ r[i] = lo[i]; r[8+i] = hi[i]; }
  return r;
}

static __device__ __forceinline__ v8f wmma_f16(v16h a, v16h b, v8f c){
  // D = A(16x32 f16) * B(32x16 f16) + C(16x16 f32)
  return __builtin_amdgcn_wmma_f32_16x16x32_f16(false, a, false, b, (short)0, c,
                                                false, false);
}

// ---------------------------------------------------------------- K0: weights f32->f16
__global__ void cvt_weights(const float* __restrict__ w1, const float* __restrict__ w2,
                            _Float16* __restrict__ w1h, _Float16* __restrict__ w2h){
  const int n1 = C_M * K_I2C;
  const int n2 = C_MASK * C_M;
  int stride = gridDim.x * blockDim.x;
  for (int t = blockIdx.x * blockDim.x + threadIdx.x; t < n1 + n2; t += stride){
    if (t < n1) w1h[t] = (_Float16)w1[t];
    else        w2h[t - n1] = (_Float16)w2[t - n1];
  }
}

// ---------------------------------------------------------------- K1: flow conv1 (128->2, 3x3)
__global__ void fh_conv1(const float* __restrict__ img, const float* __restrict__ w,
                         const float* __restrict__ b, float* __restrict__ hpre){
  int t = blockIdx.x * blockDim.x + threadIdx.x;
  if (t >= 2 * MPIX) return;
  int c = t >> 14;
  int m = t & (MPIX - 1);
  int n = m >> 12, h = (m >> 6) & 63, x = m & 63;
  float acc = b[c];
  for (int ci = 0; ci < C_IN; ++ci){
    const float* ip = img + (size_t)(n * C_IN + ci) * HW;
    const float* wp = w + (c * C_IN + ci) * 9;
#pragma unroll
    for (int di = 0; di < 3; ++di){
      int hh = h + di - 1;
      if ((unsigned)hh >= 64u) continue;
#pragma unroll
      for (int dj = 0; dj < 3; ++dj){
        int wy = x + dj - 1;
        if ((unsigned)wy >= 64u) continue;
        acc = fmaf(ip[hh * 64 + wy], wp[di * 3 + dj], acc);
      }
    }
  }
  hpre[(size_t)c * MPIX + m] = acc;
}

// ---------------------------------------------------------------- K2/K6: BN stats -> scale/shift
__global__ void bn_stats(const float* __restrict__ x, const float* __restrict__ g,
                         const float* __restrict__ beta, float* __restrict__ scale,
                         float* __restrict__ shift){
  __shared__ float s1[256], s2[256];
  int c = blockIdx.x;
  const float* p = x + (size_t)c * MPIX;
  float a = 0.f, b = 0.f;
  for (int i = threadIdx.x; i < MPIX; i += 256){ float v = p[i]; a += v; b += v * v; }
  s1[threadIdx.x] = a; s2[threadIdx.x] = b;
  __syncthreads();
  for (int o = 128; o > 0; o >>= 1){
    if ((int)threadIdx.x < o){
      s1[threadIdx.x] += s1[threadIdx.x + o];
      s2[threadIdx.x] += s2[threadIdx.x + o];
    }
    __syncthreads();
  }
  if (threadIdx.x == 0){
    float mu  = s1[0] * (1.0f / MPIX);
    float var = s2[0] * (1.0f / MPIX) - mu * mu;
    float r   = rsqrtf(var + 1e-5f);
    float sc  = r * g[c];
    scale[c] = sc;
    shift[c] = beta[c] - mu * sc;
  }
}

// ---------------------------------------------------------------- K3: flow conv2 (2->2, 3x3) on bn+relu
__global__ void fh_conv2(const float* __restrict__ hpre, const float* __restrict__ scale,
                         const float* __restrict__ shift, const float* __restrict__ w,
                         const float* __restrict__ b, float* __restrict__ flow){
  int t = blockIdx.x * blockDim.x + threadIdx.x;
  if (t >= 2 * MPIX) return;
  int c = t >> 14;
  int m = t & (MPIX - 1);
  int n = m >> 12, h = (m >> 6) & 63, x = m & 63;
  float acc = b[c];
#pragma unroll
  for (int ci = 0; ci < 2; ++ci){
    float sc = scale[ci], sh = shift[ci];
    const float* ip = hpre + (size_t)ci * MPIX + n * HW;
    const float* wp = w + (c * 2 + ci) * 9;
#pragma unroll
    for (int di = 0; di < 3; ++di){
      int hh = h + di - 1;
      if ((unsigned)hh >= 64u) continue;
#pragma unroll
      for (int dj = 0; dj < 3; ++dj){
        int wy = x + dj - 1;
        if ((unsigned)wy >= 64u) continue;
        float v = fmaxf(ip[hh * 64 + wy] * sc + sh, 0.f);
        acc = fmaf(v, wp[di * 3 + dj], acc);
      }
    }
  }
  flow[(size_t)c * MPIX + m] = acc;
}

// ---------------------------------------------------------------- K4: mask conv1 (128->256, 3x3) WMMA GEMM
// D[o][m] tile (16 out-ch rows x 16 pixel cols), K = 1152 im2col, A = weights, B = im2col.
__global__ __launch_bounds__(128) void mask_conv1(const float* __restrict__ img,
                                                  const _Float16* __restrict__ w1h,
                                                  const float* __restrict__ b1,
                                                  float* __restrict__ mpre){
  __shared__ __align__(16) _Float16 Bt[16][1160]; // [pixel][k], padded stride
  int m0 = blockIdx.x * 16;
  int n = m0 >> 12, h = (m0 >> 6) & 63, w0 = m0 & 63;

  // Build im2col tile in LDS (f16), coalesced over pixels.
  for (int idx = threadIdx.x; idx < 16 * K_I2C; idx += 128){
    int ml = idx & 15;
    int k  = idx >> 4;
    int c  = k / 9;
    int r  = k - c * 9;
    int di = r / 3, dj = r - di * 3;
    int hh = h + di - 1;
    int wy = w0 + ml + dj - 1;
    float v = 0.f;
    if ((unsigned)hh < 64u && (unsigned)wy < 64u)
      v = img[((size_t)(n * C_IN + c) * 64 + hh) * 64 + wy];
    Bt[ml][k] = (_Float16)v;
  }
  __syncthreads();

  int lane = threadIdx.x & 31;
  int wave = threadIdx.x >> 5;
  int lh   = lane >> 4;   // half-wave select
  int l16  = lane & 15;

  for (int ot = wave; ot < 16; ot += 4){
    int o0 = ot * 16;
    v8f acc{};
    // A operand (chunked K layout): K = 16*(e/8) + 8*lh + e%8
    const _Float16* arow = w1h + (size_t)(o0 + l16) * K_I2C + 8 * lh;
    // B operand (chunked K layout): K = k0 + 16*(e/8) + 8*lh + e%8
    const _Float16* brow = &Bt[l16][8 * lh];
#pragma unroll 4
    for (int ks = 0; ks < 36; ++ks){
      int k0 = ks * 32;
      v8h alo = *(const v8h*)(arow + k0);
      v8h ahi = *(const v8h*)(arow + k0 + 16);
      v8h blo = *(const v8h*)(brow + k0);
      v8h bhi = *(const v8h*)(brow + k0 + 16);
      acc = wmma_f16(pack16(alo, ahi), pack16(blo, bhi), acc);
    }
#pragma unroll
    for (int r = 0; r < 8; ++r){
      int o = o0 + r + 8 * lh;         // D row = out channel
      mpre[(size_t)o * MPIX + m0 + l16] = acc[r] + b1[o]; // D col = pixel (coalesced)
    }
  }
}

// ---------------------------------------------------------------- K7: mask conv2 (1x1, 256->2304) WMMA
// + fused bias, *0.25, softmax over 9 taps, convex-combine with 8*flow, final layout write.
__global__ __launch_bounds__(128) void mask_conv2(const float* __restrict__ mpre,
                                                  const float* __restrict__ scale,
                                                  const float* __restrict__ shift,
                                                  const _Float16* __restrict__ w2h,
                                                  const float* __restrict__ b2,
                                                  const float* __restrict__ flow,
                                                  float* __restrict__ out){
  __shared__ __align__(16) _Float16 At[16][264];  // [pixel][channel] bn+relu f16
  __shared__ float flowT[2][3][18];               // 8*flow 3x3 halo for this pixel row-tile
  int m0 = blockIdx.x * 16;
  int n = m0 >> 12, h = (m0 >> 6) & 63, w0 = m0 & 63;

  for (int o = threadIdx.x; o < C_M; o += 128){
    float sc = scale[o], sh = shift[o];
    const float* p = mpre + (size_t)o * MPIX + m0;
#pragma unroll
    for (int px = 0; px < 16; ++px)
      At[px][o] = (_Float16)fmaxf(p[px] * sc + sh, 0.f);
  }
  if (threadIdx.x < 108){
    int c   = threadIdx.x / 54;
    int rem = threadIdx.x - c * 54;
    int di  = rem / 18;
    int x   = rem - di * 18;
    int hh  = h + di - 1;
    int wy  = w0 + x - 1;
    float v = 0.f;
    if ((unsigned)hh < 64u && (unsigned)wy < 64u)
      v = 8.0f * flow[(size_t)c * MPIX + (n * 64 + hh) * 64 + wy];
    flowT[c][di][x] = v;
  }
  __syncthreads();

  int lane = threadIdx.x & 31;
  int wave = threadIdx.x >> 5;
  int lh   = lane >> 4;
  int l16  = lane & 15;
  const _Float16* arow = &At[l16][8 * lh];

  for (int q = 0; q < 4; ++q){
    int rt = wave * 4 + q;            // i index (0..15) of the 16x16 sub-pixel grid
    v8f acc[9];
#pragma unroll
    for (int k = 0; k < 9; ++k){
      v8f cacc{};
      const _Float16* brow = w2h + (size_t)(k * C_M + rt * 16 + l16) * C_M + 8 * lh;
#pragma unroll
      for (int ks = 0; ks < 8; ++ks){
        int c0 = ks * 32;
        v8h alo = *(const v8h*)(arow + c0);
        v8h ahi = *(const v8h*)(arow + c0 + 16);
        v8h blo = *(const v8h*)(brow + c0);
        v8h bhi = *(const v8h*)(brow + c0 + 16);
        cacc = wmma_f16(pack16(alo, ahi), pack16(blo, bhi), cacc);
      }
      acc[k] = cacc;
    }
    float bb[9];
#pragma unroll
    for (int k = 0; k < 9; ++k) bb[k] = b2[k * C_M + rt * 16 + l16];
#pragma unroll
    for (int r = 0; r < 8; ++r){
      int p = r + 8 * lh;             // pixel within tile
      float x[9];
      float mx = -3.0e38f;
#pragma unroll
      for (int k = 0; k < 9; ++k){ x[k] = 0.25f * (acc[k][r] + bb[k]); mx = fmaxf(mx, x[k]); }
      float s = 0.f;
#pragma unroll
      for (int k = 0; k < 9; ++k){ x[k] = __expf(x[k] - mx); s += x[k]; }
      float inv = 1.0f / s;
      float o0v = 0.f, o1v = 0.f;
#pragma unroll
      for (int k = 0; k < 9; ++k){
        float pk = x[k] * inv;
        int di = k / 3, dj = k - di * 3;
        o0v = fmaf(pk, flowT[0][di][p + dj], o0v);
        o1v = fmaf(pk, flowT[1][di][p + dj], o1v);
      }
      int row = h * 16 + rt;
      int col = (w0 + p) * 16 + l16;  // lanes 0..15 -> contiguous 64B store
      size_t base = ((size_t)n * 2 * 1024 + row) * 1024 + col;
      out[base] = o0v;
      out[base + (size_t)1024 * 1024] = o1v;
    }
  }
}

// ----------------------------------------------------------------
extern "C" void kernel_launch(void* const* d_in, const int* in_sizes, int n_in,
                              void* d_out, int out_size, void* d_ws, size_t ws_size,
                              hipStream_t stream){
  const float* img   = (const float*)d_in[0];
  const float* fh_w1 = (const float*)d_in[1];
  const float* fh_b1 = (const float*)d_in[2];
  const float* fh_g  = (const float*)d_in[3];
  const float* fh_be = (const float*)d_in[4];
  const float* fh_w2 = (const float*)d_in[5];
  const float* fh_b2 = (const float*)d_in[6];
  const float* mk_w1 = (const float*)d_in[7];
  const float* mk_b1 = (const float*)d_in[8];
  const float* mk_g  = (const float*)d_in[9];
  const float* mk_be = (const float*)d_in[10];
  const float* mk_w2 = (const float*)d_in[11];
  const float* mk_b2 = (const float*)d_in[12];
  float* out = (float*)d_out;

  char* ws = (char*)d_ws;
  auto take = [&](size_t bytes) -> char* {
    char* p = ws; ws += (bytes + 255) & ~(size_t)255; return p;
  };
  _Float16* W1h = (_Float16*)take(sizeof(_Float16) * C_M * K_I2C);      // 576 KB
  _Float16* W2h = (_Float16*)take(sizeof(_Float16) * C_MASK * C_M);     // 1.13 MB
  float* mpre = (float*)take(sizeof(float) * (size_t)C_M * MPIX);       // 16 MB
  float* hpre = (float*)take(sizeof(float) * 2 * MPIX);
  float* flw  = (float*)take(sizeof(float) * 2 * MPIX);
  float* scF  = (float*)take(sizeof(float) * 2);
  float* shF  = (float*)take(sizeof(float) * 2);
  float* scM  = (float*)take(sizeof(float) * C_M);
  float* shM  = (float*)take(sizeof(float) * C_M);

  cvt_weights<<<432, 256, 0, stream>>>(mk_w1, mk_w2, W1h, W2h);
  fh_conv1<<<(2 * MPIX) / 256, 256, 0, stream>>>(img, fh_w1, fh_b1, hpre);
  bn_stats<<<2, 256, 0, stream>>>(hpre, fh_g, fh_be, scF, shF);
  fh_conv2<<<(2 * MPIX) / 256, 256, 0, stream>>>(hpre, scF, shF, fh_w2, fh_b2, flw);
  mask_conv1<<<MPIX / 16, 128, 0, stream>>>(img, W1h, mk_b1, mpre);
  bn_stats<<<C_M, 256, 0, stream>>>(mpre, mk_g, mk_be, scM, shM);
  mask_conv2<<<MPIX / 16, 128, 0, stream>>>(mpre, scM, shM, W2h, mk_b2, flw, out);

  (void)in_sizes; (void)n_in; (void)out_size; (void)ws_size;
}
